// MultiHeadAttention_53197464928551
// MI455X (gfx1250) — matrix operand order repaired and checked by reading, not compiled, not run on hardware
//
#include <hip/hip_runtime.h>
#include <stdint.h>

#define BATCH 4
#define SEQ   2048
#define DIM   1024
#define NH    16
#define HD    64
#define MTOT  (BATCH * SEQ)

typedef __attribute__((ext_vector_type(16))) __bf16 v16bf;
typedef __attribute__((ext_vector_type(8)))  float  v8f;
typedef uint32_t u32x4 __attribute__((ext_vector_type(4)));
typedef uint32_t u32x8 __attribute__((ext_vector_type(8)));
typedef __attribute__((address_space(3))) unsigned short lds_u16_t;

union FragAB { v16bf v; uint4 q[2]; };
union FragC  { v8f   v; float f[8]; };

__device__ __forceinline__ uint32_t f2bf(float x) {
  uint32_t u = __float_as_uint(x);
  return (u + 0x7fffu + ((u >> 16) & 1u)) >> 16;   // RNE truncate to bf16
}
__device__ __forceinline__ uint32_t packbf(float lo, float hi) {
  return f2bf(lo) | (f2bf(hi) << 16);
}

__device__ __forceinline__ void pack_store16(const float4 f[4], unsigned short* dst) {
  uint4 p0, p1;
  p0.x = packbf(f[0].x, f[0].y); p0.y = packbf(f[0].z, f[0].w);
  p0.z = packbf(f[1].x, f[1].y); p0.w = packbf(f[1].z, f[1].w);
  p1.x = packbf(f[2].x, f[2].y); p1.y = packbf(f[2].z, f[2].w);
  p1.z = packbf(f[3].x, f[3].y); p1.w = packbf(f[3].z, f[3].w);
  uint4* d = (uint4*)dst;
  d[0] = p0; d[1] = p1;
}

// =====================================================================
// GEMM: C[M,N] = A[M,K] * B[N,K]^T + bias   (K = N = DIM, M = MTOT)
// double-buffered LDS, one barrier per k-step.
// mode 0: store f32 row-major [M,N]
// mode 1: store bf16 per-head layout [B,H,L,HD]  (N -> head*64+hd)
// =====================================================================
#define BM 128
#define BN 128
#define BK 32
#define TPAD 40   // padded LDS row stride (elements)

__global__ __launch_bounds__(256)
void mha_gemm_bf16_kernel(const float* __restrict__ A, const float* __restrict__ Bm,
                          const float* __restrict__ bias, void* __restrict__ outp,
                          int mode) {
  __shared__ __align__(16) unsigned short As[2][BM * TPAD];
  __shared__ __align__(16) unsigned short Bs[2][BN * TPAD];

  const int t    = threadIdx.x;
  const int lane = t & 31;
  const int wv   = t >> 5;
  const int r    = lane & 15;
  const int hh   = lane >> 4;
  const int wm   = (wv & 3) * 32;   // 4 waves along M
  const int wn   = (wv >> 2) * 64;  // 2 waves along N

  const int mbase = blockIdx.y * BM;
  const int nbase = blockIdx.x * BN;

  const int lrow = t >> 1;          // 0..127
  const int lcol = (t & 1) * 16;    // 0 or 16 floats
  const float* gA = A  + (size_t)(mbase + lrow) * DIM + lcol;
  const float* gB = Bm + (size_t)(nbase + lrow) * DIM + lcol;
  const int ldst  = lrow * TPAD + lcol;

  FragC acc[2][4];
#pragma unroll
  for (int mi = 0; mi < 2; ++mi)
#pragma unroll
    for (int ni = 0; ni < 4; ++ni)
#pragma unroll
      for (int i = 0; i < 8; ++i) acc[mi][ni].f[i] = 0.0f;

  float4 fa[4], fb[4];
  { // prologue: tile 0 into buffer 0
    const float4* pa = (const float4*)gA;
    const float4* pb = (const float4*)gB;
    fa[0] = pa[0]; fa[1] = pa[1]; fa[2] = pa[2]; fa[3] = pa[3];
    fb[0] = pb[0]; fb[1] = pb[1]; fb[2] = pb[2]; fb[3] = pb[3];
    pack_store16(fa, &As[0][ldst]);
    pack_store16(fb, &Bs[0][ldst]);
  }
  __syncthreads();

  int buf = 0;
  for (int kb = 0; kb < DIM; kb += BK, buf ^= 1) {
    const bool has_next = (kb + BK) < DIM;
    if (has_next) { // issue next tile's global loads early
      const float4* pa = (const float4*)(gA + kb + BK);
      const float4* pb = (const float4*)(gB + kb + BK);
      fa[0] = pa[0]; fa[1] = pa[1]; fa[2] = pa[2]; fa[3] = pa[3];
      fb[0] = pb[0]; fb[1] = pb[1]; fb[2] = pb[2]; fb[3] = pb[3];
    }

    // fragments for current buffer, then 8 back-to-back WMMA
    FragAB a[2];
#pragma unroll
    for (int mi = 0; mi < 2; ++mi) {
      const unsigned short* ar = &As[buf][(wm + mi * 16 + r) * TPAD];
      a[mi].q[0] = *(const uint4*)(ar + hh * 8);
      a[mi].q[1] = *(const uint4*)(ar + 16 + hh * 8);
    }
    FragAB bfr[4];
#pragma unroll
    for (int ni = 0; ni < 4; ++ni) {
      const unsigned short* br = &Bs[buf][(wn + ni * 16 + r) * TPAD + hh * 8];
      bfr[ni].q[0] = *(const uint4*)(br);
      bfr[ni].q[1] = *(const uint4*)(br + 16);
    }
#pragma unroll
    for (int ni = 0; ni < 4; ++ni)
#pragma unroll
      for (int mi = 0; mi < 2; ++mi)
        acc[mi][ni].v = __builtin_amdgcn_wmma_f32_16x16x32_bf16(
            false, a[mi].v, false, bfr[ni].v, (short)0, acc[mi][ni].v, false, false);

    if (has_next) { // stage next tile into the other buffer
      pack_store16(fa, &As[buf ^ 1][ldst]);
      pack_store16(fb, &Bs[buf ^ 1][ldst]);
    }
    __syncthreads();
  }

  // epilogue: C element (M = hh*8+i, N = r)
#pragma unroll
  for (int ni = 0; ni < 4; ++ni) {
    const int n  = nbase + wn + ni * 16 + r;
    const float bvv = bias[n];
#pragma unroll
    for (int mi = 0; mi < 2; ++mi) {
#pragma unroll
      for (int i = 0; i < 8; ++i) {
        const int m = mbase + wm + mi * 16 + hh * 8 + i;
        const float val = acc[mi][ni].f[i] + bvv;
        if (mode == 0) {
          ((float*)outp)[(size_t)m * DIM + n] = val;
        } else {
          const int bb   = m >> 11;          // m / SEQ
          const int l    = m & (SEQ - 1);
          const int head = n >> 6;           // n / HD
          const int hd   = n & (HD - 1);
          ((unsigned short*)outp)[((((size_t)bb * NH + head) * SEQ) + l) * HD + hd] =
              (unsigned short)f2bf(val);
        }
      }
    }
  }
}

// =====================================================================
// Flash attention: bf16 Q/K/V in [B,H,L,64], f32 output [B,L,D].
// 8 waves x 16 query rows; K tile streamed by the Tensor Data Mover
// directly into padded LDS; V transposed through registers.
// =====================================================================
#define QT 128
#define KT 32
#define QPAD 72
#define KPAD 72   // 64 data + 8 pad elems = 128B + 16B  -> TDM pad 4 DW / 32 DW
#define VPAD 40
#define PPAD 40

__global__ __launch_bounds__(256)
void mha_flash_attn_kernel(const unsigned short* __restrict__ Q,
                           const unsigned short* __restrict__ K,
                           const unsigned short* __restrict__ V,
                           float* __restrict__ Obuf) {
  __shared__ __align__(16) unsigned short Qs[QT * QPAD];
  __shared__ __align__(16) unsigned short Ks[KT * KPAD];
  __shared__ __align__(16) unsigned short Vts[HD * VPAD];      // V^T [hd][key]
  __shared__ __align__(16) unsigned short Ps[8 * 16 * PPAD];   // per-wave P

  const int t    = threadIdx.x;
  const int lane = t & 31;
  const int wv   = t >> 5;
  const int r    = lane & 15;
  const int hh   = lane >> 4;

  const int bh = blockIdx.y;
  const int q0 = blockIdx.x * QT;
  const size_t headbase = (size_t)bh * SEQ * HD;

  { // Q tile: 128 x 64 bf16
    const int row = t >> 1;
    const int col = (t & 1) * 32;
    const uint4* gq = (const uint4*)(Q + headbase + (size_t)(q0 + row) * HD + col);
    uint4* dq = (uint4*)&Qs[row * QPAD + col];
    dq[0] = gq[0]; dq[1] = gq[1]; dq[2] = gq[2]; dq[3] = gq[3];
  }

  const unsigned ldsK = (unsigned)(unsigned long long)(lds_u16_t*)&Ks[0];

  FragC o[4];
  float mrow[8], lrow[8];
#pragma unroll
  for (int nt = 0; nt < 4; ++nt)
#pragma unroll
    for (int i = 0; i < 8; ++i) o[nt].f[i] = 0.0f;
#pragma unroll
  for (int i = 0; i < 8; ++i) { mrow[i] = -1e30f; lrow[i] = 0.0f; }

  const float scale = 0.125f;   // 1/sqrt(64)

  for (int k0 = 0; k0 < SEQ; k0 += KT) {
    if (wv == 0) {
      // TDM: DMA 32x64 bf16 K tile -> Ks, padding 16B after every 128B row.
      const uint64_t ga = (uint64_t)(const void*)(K + headbase + (size_t)k0 * HD);
      u32x4 g0;
      g0[0] = 1u;                                              // count=1
      g0[1] = ldsK;                                            // lds_addr
      g0[2] = (uint32_t)ga;                                    // global_addr lo
      g0[3] = ((uint32_t)(ga >> 32) & 0x01FFFFFFu) | (2u << 30); // addr hi | type=2
      u32x8 g1;
      g1[0] = (1u << 16) | (1u << 20) | (4u << 22) | (3u << 25); // 2B, pad on, 32DW/4DW
      g1[1] = (uint32_t)HD << 16;                                // tensor_dim0 = 64
      g1[2] = ((uint32_t)SEQ & 0xFFFFu) << 16;                   // tensor_dim1 = 2048
      g1[3] = (uint32_t)HD << 16;                                // tile_dim0 = 64
      g1[4] = (uint32_t)KT;                                      // tile_dim1 = 32
      g1[5] = (uint32_t)HD;                                      // dim0_stride = 64
      g1[6] = 0u;
      g1[7] = 0u;
      asm volatile("tensor_load_to_lds %0, %1" :: "s"(g0), "s"(g1) : "memory");
    }
    { // V tile: load + transpose into Vts
      const int row = t >> 3;        // 0..31
      const int col = (t & 7) * 8;   // 0..56
      const size_t gbase = headbase + (size_t)(k0 + row) * HD + col;
      union { uint4 q; unsigned short s[8]; } vd;
      vd.q = *(const uint4*)(V + gbase);
#pragma unroll
      for (int j = 0; j < 8; ++j) Vts[(col + j) * VPAD + row] = vd.s[j];
      if (k0 + KT < SEQ)
        __builtin_prefetch(V + gbase + (size_t)KT * HD, 0, 1);
    }
    __builtin_amdgcn_s_wait_tensorcnt(0);
    __syncthreads();

    // ---- S = Q K^T (16 x 32 per wave) ----
    FragAB aq[2];
#pragma unroll
    for (int ks = 0; ks < 2; ++ks) {
      const unsigned short* qr = &Qs[(wv * 16 + r) * QPAD + ks * 32];
      aq[ks].q[0] = *(const uint4*)(qr + hh * 8);
      aq[ks].q[1] = *(const uint4*)(qr + 16 + hh * 8);
    }
    FragAB bk[2][2];
#pragma unroll
    for (int nt = 0; nt < 2; ++nt)
#pragma unroll
      for (int ks = 0; ks < 2; ++ks) {
        const unsigned short* kr = &Ks[(nt * 16 + r) * KPAD + ks * 32 + hh * 8];
        bk[nt][ks].q[0] = *(const uint4*)(kr);
        bk[nt][ks].q[1] = *(const uint4*)(kr + 16);
      }
    FragC s[2];
#pragma unroll
    for (int nt = 0; nt < 2; ++nt)
#pragma unroll
      for (int i = 0; i < 8; ++i) s[nt].f[i] = 0.0f;
#pragma unroll
    for (int ks = 0; ks < 2; ++ks)
#pragma unroll
      for (int nt = 0; nt < 2; ++nt)
        s[nt].v = __builtin_amdgcn_wmma_f32_16x16x32_bf16(
            false, aq[ks].v, false, bk[nt][ks].v, (short)0, s[nt].v, false, false);

    // ---- online softmax: row M=hh*8+i spread over 16 lanes ----
#pragma unroll
    for (int i = 0; i < 8; ++i) {
      float s0 = s[0].f[i] * scale;
      float s1 = s[1].f[i] * scale;
      float mx = fmaxf(s0, s1);
#pragma unroll
      for (int msk = 1; msk < 16; msk <<= 1)
        mx = fmaxf(mx, __shfl_xor(mx, msk, 32));
      const float mnew  = fmaxf(mrow[i], mx);
      const float alpha = __expf(mrow[i] - mnew);
      const float p0 = __expf(s0 - mnew);
      const float p1 = __expf(s1 - mnew);
      float rs = p0 + p1;
#pragma unroll
      for (int msk = 1; msk < 16; msk <<= 1)
        rs += __shfl_xor(rs, msk, 32);
      lrow[i] = lrow[i] * alpha + rs;
      mrow[i] = mnew;
#pragma unroll
      for (int nt = 0; nt < 4; ++nt) o[nt].f[i] *= alpha;
      unsigned short* pr = &Ps[(wv * 16 + hh * 8 + i) * PPAD];
      pr[r]      = (unsigned short)f2bf(p0);
      pr[16 + r] = (unsigned short)f2bf(p1);
    }
    __syncthreads();

    // ---- O += P (16x32) * V (32x64) ----
    {
      const unsigned short* pr = &Ps[(wv * 16 + r) * PPAD];
      FragAB ap;
      ap.q[0] = *(const uint4*)(pr + hh * 8);
      ap.q[1] = *(const uint4*)(pr + 16 + hh * 8);
      FragAB bv[4];
#pragma unroll
      for (int nt = 0; nt < 4; ++nt) {
        const unsigned short* vr = &Vts[(nt * 16 + r) * VPAD + hh * 8];
        bv[nt].q[0] = *(const uint4*)(vr);
        bv[nt].q[1] = *(const uint4*)(vr + 16);
      }
#pragma unroll
      for (int nt = 0; nt < 4; ++nt)
        o[nt].v = __builtin_amdgcn_wmma_f32_16x16x32_bf16(
            false, ap.v, false, bv[nt].v, (short)0, o[nt].v, false, false);
    }
    __syncthreads();
  }

  // ---- normalize + store f32 [B,L,D] ----
  const int b    = bh >> 4;
  const int head = bh & (NH - 1);
  float inv[8];
#pragma unroll
  for (int i = 0; i < 8; ++i) inv[i] = 1.0f / lrow[i];
#pragma unroll
  for (int nt = 0; nt < 4; ++nt) {
#pragma unroll
    for (int i = 0; i < 8; ++i) {
      const int l   = q0 + wv * 16 + hh * 8 + i;
      const int col = head * HD + nt * 16 + r;
      Obuf[((size_t)b * SEQ + l) * DIM + col] = o[nt].f[i] * inv[i];
    }
  }
}

// =====================================================================
extern "C" void kernel_launch(void* const* d_in, const int* in_sizes, int n_in,
                              void* d_out, int out_size, void* d_ws, size_t ws_size,
                              hipStream_t stream) {
  const float* x  = (const float*)d_in[0];
  const float* Wq = (const float*)d_in[1];
  const float* bq = (const float*)d_in[2];
  const float* Wk = (const float*)d_in[3];
  const float* bk = (const float*)d_in[4];
  const float* Wv = (const float*)d_in[5];
  const float* bv = (const float*)d_in[6];
  const float* Wo = (const float*)d_in[7];
  const float* bo = (const float*)d_in[8];
  float* out = (float*)d_out;

  char* ws = (char*)d_ws;
  unsigned short* Qbf = (unsigned short*)(ws);                        // 16 MiB
  unsigned short* Kbf = (unsigned short*)(ws + ((size_t)16 << 20));   // 16 MiB
  unsigned short* Vbf = (unsigned short*)(ws + ((size_t)32 << 20));   // 16 MiB
  float*          Obf = (float*)(ws + ((size_t)48 << 20));            // 32 MiB

  dim3 blk(256);
  dim3 ggrid(DIM / BN, MTOT / BM);   // (8, 64)

  mha_gemm_bf16_kernel<<<ggrid, blk, 0, stream>>>(x, Wq, bq, (void*)Qbf, 1);
  mha_gemm_bf16_kernel<<<ggrid, blk, 0, stream>>>(x, Wk, bk, (void*)Kbf, 1);
  mha_gemm_bf16_kernel<<<ggrid, blk, 0, stream>>>(x, Wv, bv, (void*)Vbf, 1);

  dim3 agrid(SEQ / QT, BATCH * NH);  // (16, 64)
  mha_flash_attn_kernel<<<agrid, blk, 0, stream>>>(Qbf, Kbf, Vbf, Obf);

  mha_gemm_bf16_kernel<<<ggrid, blk, 0, stream>>>(Obf, Wo, bo, (void*)out, 0);
}
